// DL_GATv2_39908836115049
// MI455X (gfx1250) — matrix operand order repaired and checked by reading, not compiled, not run on hardware
//
#include <hip/hip_runtime.h>
#include <hip/hip_bf16.h>

typedef __attribute__((ext_vector_type(16))) _Float16 v16h;
typedef __attribute__((ext_vector_type(8)))  _Float16 v8h;
typedef __attribute__((ext_vector_type(8)))  float    v8f;

#define GN   50000
#define GE   800000
#define GHC  128
#define GB   64
#define GDD  32

// ---------------------------------------------------------------------------
// f32 -> f16 conversion
// ---------------------------------------------------------------------------
__global__ void k_f32_to_f16(const float* __restrict__ in, _Float16* __restrict__ out, long n) {
    long i = (long)blockIdx.x * blockDim.x + threadIdx.x;
    if (i < n) out[i] = (_Float16)in[i];
}

// ---------------------------------------------------------------------------
// WMMA GEMM:  Y[N,128] = X[N,128](f16) @ W[128,128](f16, row = out-col)^T + bias
// One wave per 16-row tile; 8 waves / block.
// ---------------------------------------------------------------------------
__global__ __launch_bounds__(256) void k_wmma_gemm128_bias(
    const _Float16* __restrict__ X, const _Float16* __restrict__ W,
    const float* __restrict__ bias, float* __restrict__ Y, int nTiles)
{
    const int lane = threadIdx.x & 31;
    const int wave = threadIdx.x >> 5;
    const int tile = blockIdx.x * 8 + wave;
    if (tile >= nTiles) return;
    const int m  = lane & 15;     // row within tile (A) / col within ntile (B,D)
    const int hf = lane >> 4;     // lane half
    const _Float16* xrow = X + ((long)tile * 16 + m) * GHC;

    // A fragments for the 4 K-steps (K=32 each), per ISA 16-bit A layout:
    // VGPR0..3 -> K = kb + hf*8 + {0..7}; VGPR4..7 -> K = kb + 16 + hf*8 + {0..7}
    v16h afrag[4];
#pragma unroll
    for (int kt = 0; kt < 4; ++kt) {
        const int kb = kt * 32;
        v8h lo = *(const v8h*)(xrow + kb + hf * 8);
        v8h hi = *(const v8h*)(xrow + kb + 16 + hf * 8);
#pragma unroll
        for (int i = 0; i < 8; ++i) { afrag[kt][i] = lo[i]; afrag[kt][8 + i] = hi[i]; }
    }

#pragma unroll
    for (int nt = 0; nt < 8; ++nt) {
        const int col = nt * 16 + m;
        const _Float16* wrow = W + (long)col * GHC;   // weights for this output col
        v8f acc = {};
#pragma unroll
        for (int kt = 0; kt < 4; ++kt) {
            // B layout: same as A: K = kb + hf*8 + {0..7}, then kb + 16 + hf*8 + {0..7}
            v8h blo = *(const v8h*)(wrow + kt * 32 + hf * 8);
            v8h bhi = *(const v8h*)(wrow + kt * 32 + 16 + hf * 8);
            v16h b;
#pragma unroll
            for (int i = 0; i < 8; ++i) { b[i] = blo[i]; b[8 + i] = bhi[i]; }
            acc = __builtin_amdgcn_wmma_f32_16x16x32_f16(
                false, afrag[kt], false, b, (short)0, acc, false, false);
        }
        const float bv = bias[col];
        // D layout: VGPR r -> row = r + 8*hf, col = m (within tile)
        float* yout = Y + ((long)tile * 16 + 8 * hf) * GHC + col;
#pragma unroll
        for (int r = 0; r < 8; ++r) yout[(long)r * GHC] = acc[r] + bv;
    }
}

// ---------------------------------------------------------------------------
// Edge pass 1: score + per-dst segment max (monotone unsigned key atomicMax)
// ---------------------------------------------------------------------------
__device__ __forceinline__ unsigned f2key(float f) {
    unsigned u = __float_as_uint(f);
    return (u & 0x80000000u) ? ~u : (u | 0x80000000u);
}
__device__ __forceinline__ float key2f(unsigned k) {
    unsigned u = (k & 0x80000000u) ? (k & 0x7FFFFFFFu) : ~k;
    return __uint_as_float(u);
}

__global__ void k_edge_score(const int* __restrict__ srcs, const int* __restrict__ dsts,
                             const float* __restrict__ XL, const float* __restrict__ XR,
                             const float* __restrict__ att, float* __restrict__ S,
                             unsigned* __restrict__ smaxkey, long total /* E*4 */)
{
    long t = (long)blockIdx.x * blockDim.x + threadIdx.x;
    if (t >= total) return;
    const long e = t >> 2; const int hd = (int)(t & 3);
    const int src = srcs[e], dst = dsts[e];
    const float* pl = XL + (long)src * GHC + hd * 32;
    const float* pr = XR + (long)dst * GHC + hd * 32;
    const float* pa = att + hd * 32;
    float s = 0.f;
#pragma unroll
    for (int c = 0; c < 32; ++c) {
        float v = pl[c] + pr[c];
        v = v > 0.f ? v : 0.2f * v;       // leaky_relu(., 0.2)
        s += pa[c] * v;
    }
    S[t] = s;
    atomicMax(&smaxkey[(long)dst * 4 + hd], f2key(s));
}

// ---------------------------------------------------------------------------
// Edge pass 2: a = exp(s - smax[dst]);  denom[dst] += a   (a stored in-place)
// ---------------------------------------------------------------------------
__global__ void k_edge_exp(const int* __restrict__ dsts, float* __restrict__ S,
                           const unsigned* __restrict__ smaxkey,
                           float* __restrict__ denom, long total)
{
    long t = (long)blockIdx.x * blockDim.x + threadIdx.x;
    if (t >= total) return;
    const long e = t >> 2; const int hd = (int)(t & 3);
    const int dst = dsts[e];
    const float smax = key2f(smaxkey[(long)dst * 4 + hd]);
    const float a = expf(S[t] - smax);
    S[t] = a;
    atomicAdd(&denom[(long)dst * 4 + hd], a);
}

// ---------------------------------------------------------------------------
// Edge pass 3: AGG[dst] += (a/denom[dst]) * XL[src]   (thread per edge*channel)
// ---------------------------------------------------------------------------
__global__ void k_edge_agg(const int* __restrict__ srcs, const int* __restrict__ dsts,
                           const float* __restrict__ S, const float* __restrict__ denom,
                           const float* __restrict__ XL, float* __restrict__ AGG,
                           long total /* E*128 */)
{
    long t = (long)blockIdx.x * blockDim.x + threadIdx.x;
    if (t >= total) return;
    const long e = t >> 7; const int c = (int)(t & 127); const int hd = c >> 5;
    const int src = srcs[e], dst = dsts[e];
    const float alpha = S[e * 4 + hd] / denom[(long)dst * 4 + hd];
    atomicAdd(&AGG[(long)dst * GHC + c], alpha * XL[(long)src * GHC + c]);
}

// ---------------------------------------------------------------------------
// Node update: OUT = relu(AGG + bias) + RES   (elementwise; in-place safe)
// ---------------------------------------------------------------------------
__global__ void k_node_update(const float* __restrict__ AGGi, const float* __restrict__ bias,
                              const float* __restrict__ RESi, float* __restrict__ OUT, long total)
{
    long t = (long)blockIdx.x * blockDim.x + threadIdx.x;
    if (t >= total) return;
    const int c = (int)(t & 127);
    float v = AGGi[t] + bias[c];
    v = v > 0.f ? v : 0.f;
    OUT[t] = v + RESi[t];
}

// ---------------------------------------------------------------------------
// Global mean pool (sums + counts via atomics)
// ---------------------------------------------------------------------------
__global__ void k_pool(const float* __restrict__ Hn, const int* __restrict__ batch,
                       float* __restrict__ psum, float* __restrict__ pcnt, long total)
{
    long t = (long)blockIdx.x * blockDim.x + threadIdx.x;
    if (t >= total) return;
    const long node = t >> 7; const int c = (int)(t & 127);
    const int b = batch[node];
    atomicAdd(&psum[(long)b * GHC + c], Hn[t]);
    if (c == 0) atomicAdd(&pcnt[b], 1.0f);
}

// ---------------------------------------------------------------------------
// Head MLP: single block, staged with __syncthreads, scratch in workspace
// ---------------------------------------------------------------------------
__global__ __launch_bounds__(256) void k_head(
    const float* __restrict__ psum, const float* __restrict__ pcnt,
    const float* __restrict__ domain,
    const float* __restrict__ Wg, const float* __restrict__ bg,
    const float* __restrict__ Wd, const float* __restrict__ bd,
    const float* __restrict__ Wf1, const float* __restrict__ bf1,
    const float* __restrict__ Wf2, const float* __restrict__ bf2,
    const float* __restrict__ Wf3, const float* __restrict__ bf3,
    float* __restrict__ pm, float* __restrict__ gbuf, float* __restrict__ dbuf,
    float* __restrict__ z1, float* __restrict__ z2, float* __restrict__ out)
{
    const int tid = threadIdx.x;
    // pooled mean
    for (int i = tid; i < GB * GHC; i += 256) {
        int b = i >> 7;
        float cnt = pcnt[b]; cnt = cnt > 1.f ? cnt : 1.f;
        pm[i] = psum[i] / cnt;
    }
    __syncthreads();
    // g = relu(pm @ Wg^T + bg)   [64,128]
    for (int i = tid; i < GB * GHC; i += 256) {
        int b = i >> 7, o = i & 127;
        float acc = bg[o];
        for (int k = 0; k < GHC; ++k) acc += pm[b * GHC + k] * Wg[o * GHC + k];
        gbuf[i] = acc > 0.f ? acc : 0.f;
    }
    // d = relu(domain @ Wd^T + bd)   [64,64]
    for (int i = tid; i < GB * 64; i += 256) {
        int b = i >> 6, o = i & 63;
        float acc = bd[o];
        for (int k = 0; k < GDD; ++k) acc += domain[b * GDD + k] * Wd[o * GDD + k];
        dbuf[i] = acc > 0.f ? acc : 0.f;
    }
    __syncthreads();
    // z1 = relu(cat(g,d) @ Wf1^T + bf1)   Wf1 [128,192]
    for (int i = tid; i < GB * GHC; i += 256) {
        int b = i >> 7, o = i & 127;
        float acc = bf1[o];
        const float* wr = Wf1 + o * 192;
        for (int k = 0; k < 128; ++k) acc += gbuf[b * GHC + k] * wr[k];
        for (int k = 0; k < 64;  ++k) acc += dbuf[b * 64 + k] * wr[128 + k];
        z1[i] = acc > 0.f ? acc : 0.f;
    }
    __syncthreads();
    // z2 = relu(z1 @ Wf2^T + bf2)   Wf2 [64,128]
    for (int i = tid; i < GB * 64; i += 256) {
        int b = i >> 6, o = i & 63;
        float acc = bf2[o];
        for (int k = 0; k < 128; ++k) acc += z1[b * GHC + k] * Wf2[o * 128 + k];
        z2[i] = acc > 0.f ? acc : 0.f;
    }
    __syncthreads();
    // out = z2 @ Wf3^T + bf3   [64]
    for (int b = tid; b < GB; b += 256) {
        float acc = bf3[0];
        for (int k = 0; k < 64; ++k) acc += z2[b * 64 + k] * Wf3[k];
        out[b] = acc;
    }
}

// ---------------------------------------------------------------------------
// Host orchestration
// ---------------------------------------------------------------------------
extern "C" void kernel_launch(void* const* d_in, const int* in_sizes, int n_in,
                              void* d_out, int out_size, void* d_ws, size_t ws_size,
                              hipStream_t stream) {
    (void)in_sizes; (void)n_in; (void)out_size; (void)ws_size;
    const float* x      = (const float*)d_in[0];
    const float* domain = (const float*)d_in[1];
    const float* Wl1 = (const float*)d_in[2];  const float* bl1 = (const float*)d_in[3];
    const float* Wr1 = (const float*)d_in[4];  const float* br1 = (const float*)d_in[5];
    const float* att1= (const float*)d_in[6];  const float* bias1=(const float*)d_in[7];
    const float* Wl2 = (const float*)d_in[8];  const float* bl2 = (const float*)d_in[9];
    const float* Wr2 = (const float*)d_in[10]; const float* br2 = (const float*)d_in[11];
    const float* att2= (const float*)d_in[12]; const float* bias2=(const float*)d_in[13];
    const float* Wres= (const float*)d_in[14]; const float* bres= (const float*)d_in[15];
    const float* Wd  = (const float*)d_in[16]; const float* bd  = (const float*)d_in[17];
    const float* Wg  = (const float*)d_in[18]; const float* bg  = (const float*)d_in[19];
    const float* Wf1 = (const float*)d_in[20]; const float* bf1 = (const float*)d_in[21];
    const float* Wf2 = (const float*)d_in[22]; const float* bf2 = (const float*)d_in[23];
    const float* Wf3 = (const float*)d_in[24]; const float* bf3 = (const float*)d_in[25];
    const int* eidx  = (const int*)d_in[26];   // [2,E]
    const int* batch = (const int*)d_in[27];   // [N]
    const int* srcs = eidx;
    const int* dsts = eidx + GE;
    float* out = (float*)d_out;

    // ---- workspace layout -------------------------------------------------
    size_t off = 0;
    char* base = (char*)d_ws;
    auto alloc = [&](size_t bytes) -> void* {
        void* p = base + off;
        off += (bytes + 255) & ~(size_t)255;
        return p;
    };
    const long NF = (long)GN * GHC;
    _Float16* X16   = (_Float16*)alloc(NF * 2);
    _Float16* Wl1h  = (_Float16*)alloc(GHC * GHC * 2);
    _Float16* Wr1h  = (_Float16*)alloc(GHC * GHC * 2);
    _Float16* Wresh = (_Float16*)alloc(GHC * GHC * 2);
    _Float16* Wl2h  = (_Float16*)alloc(GHC * GHC * 2);
    _Float16* Wr2h  = (_Float16*)alloc(GHC * GHC * 2);
    float* XL   = (float*)alloc(NF * 4);
    float* XR   = (float*)alloc(NF * 4);
    float* RES  = (float*)alloc(NF * 4);        // layer1 residual, then layer2 agg buffer
    float* AGG  = (float*)alloc(NF * 4);        // layer1 agg, then H (node features)
    float* S    = (float*)alloc((long)GE * 4 * 4);
    unsigned* SMAX = (unsigned*)alloc((long)GN * 4 * 4);
    float* DEN  = (float*)alloc((long)GN * 4 * 4);
    float* PSUM = (float*)alloc(GB * GHC * 4);
    float* PCNT = (float*)alloc(GB * 4);
    float* PM   = (float*)alloc(GB * GHC * 4);
    float* Gb   = (float*)alloc(GB * GHC * 4);
    float* Db   = (float*)alloc(GB * 64 * 4);
    float* Z1   = (float*)alloc(GB * GHC * 4);
    float* Z2   = (float*)alloc(GB * 64 * 4);

    const int T = 256;
    const long eh  = (long)GE * 4;    // edge*head work items
    const long ec  = (long)GE * GHC;  // edge*channel work items
    const int nTiles = GN / 16;       // 3125
    dim3 gemmGrid((nTiles + 7) / 8);
    auto blocks = [&](long n) { return dim3((unsigned)((n + T - 1) / T)); };

    // ---- convert inputs to f16 -------------------------------------------
    k_f32_to_f16<<<blocks(NF), T, 0, stream>>>(x, X16, NF);
    k_f32_to_f16<<<blocks(GHC*GHC), T, 0, stream>>>(Wl1,  Wl1h,  GHC*GHC);
    k_f32_to_f16<<<blocks(GHC*GHC), T, 0, stream>>>(Wr1,  Wr1h,  GHC*GHC);
    k_f32_to_f16<<<blocks(GHC*GHC), T, 0, stream>>>(Wres, Wresh, GHC*GHC);
    k_f32_to_f16<<<blocks(GHC*GHC), T, 0, stream>>>(Wl2,  Wl2h,  GHC*GHC);
    k_f32_to_f16<<<blocks(GHC*GHC), T, 0, stream>>>(Wr2,  Wr2h,  GHC*GHC);

    // ---- layer 1 ----------------------------------------------------------
    hipMemsetAsync(AGG,  0, NF * 4, stream);
    hipMemsetAsync(SMAX, 0, (long)GN * 4 * 4, stream);   // key 0 == -inf
    hipMemsetAsync(DEN,  0, (long)GN * 4 * 4, stream);

    k_wmma_gemm128_bias<<<gemmGrid, T, 0, stream>>>(X16, Wl1h,  bl1,  XL,  nTiles);
    k_wmma_gemm128_bias<<<gemmGrid, T, 0, stream>>>(X16, Wr1h,  br1,  XR,  nTiles);
    k_wmma_gemm128_bias<<<gemmGrid, T, 0, stream>>>(X16, Wresh, bres, RES, nTiles);

    k_edge_score<<<blocks(eh), T, 0, stream>>>(srcs, dsts, XL, XR, att1, S, SMAX, eh);
    k_edge_exp  <<<blocks(eh), T, 0, stream>>>(dsts, S, SMAX, DEN, eh);
    k_edge_agg  <<<blocks(ec), T, 0, stream>>>(srcs, dsts, S, DEN, XL, AGG, ec);
    k_node_update<<<blocks(NF), T, 0, stream>>>(AGG, bias1, RES, AGG, NF);  // AGG := H1

    // ---- layer 2 ----------------------------------------------------------
    k_f32_to_f16<<<blocks(NF), T, 0, stream>>>(AGG, X16, NF);
    hipMemsetAsync(RES,  0, NF * 4, stream);             // reuse RES as agg2
    hipMemsetAsync(SMAX, 0, (long)GN * 4 * 4, stream);
    hipMemsetAsync(DEN,  0, (long)GN * 4 * 4, stream);

    k_wmma_gemm128_bias<<<gemmGrid, T, 0, stream>>>(X16, Wl2h, bl2, XL, nTiles);
    k_wmma_gemm128_bias<<<gemmGrid, T, 0, stream>>>(X16, Wr2h, br2, XR, nTiles);

    k_edge_score<<<blocks(eh), T, 0, stream>>>(srcs, dsts, XL, XR, att2, S, SMAX, eh);
    k_edge_exp  <<<blocks(eh), T, 0, stream>>>(dsts, S, SMAX, DEN, eh);
    k_edge_agg  <<<blocks(ec), T, 0, stream>>>(srcs, dsts, S, DEN, XL, RES, ec);
    k_node_update<<<blocks(NF), T, 0, stream>>>(RES, bias2, AGG, AGG, NF);  // AGG := H2

    // ---- pool + head ------------------------------------------------------
    hipMemsetAsync(PSUM, 0, GB * GHC * 4, stream);
    hipMemsetAsync(PCNT, 0, GB * 4, stream);
    k_pool<<<blocks(NF), T, 0, stream>>>(AGG, batch, PSUM, PCNT, NF);

    k_head<<<1, 256, 0, stream>>>(PSUM, PCNT, domain, Wg, bg, Wd, bd,
                                  Wf1, bf1, Wf2, bf2, Wf3, bf3,
                                  PM, Gb, Db, Z1, Z2, out);
}
